// CBGTNet_44650480009701
// MI455X (gfx1250) — matrix operand order repaired and checked by reading, not compiled, not run on hardware
//
#include <hip/hip_runtime.h>
#include <hip/hip_bf16.h>
#include <math.h>

// ----- model dims (from reference) -----
constexpr int Bn   = 64;
constexpr int N0v  = 400;
constexpr int N1v  = 280;
constexpr int Dm   = 400;
constexpr int NH   = 4;
constexpr int DHd  = 100;
constexpr int HIDm = 800;
constexpr int DHP  = 128;
constexpr int TPAD = 416;
constexpr float ATT_SCALE = 0.1f;
constexpr long long SSTR = (long long)TPAD * TPAD;

typedef __attribute__((ext_vector_type(16))) _Float16 v16h;
typedef __attribute__((ext_vector_type(8)))  _Float16 v8h;
typedef __attribute__((ext_vector_type(8)))  float    v8f;

__device__ __forceinline__ float wsum(float v) {
#pragma unroll
  for (int o = 16; o >= 1; o >>= 1) v += __shfl_xor(v, o, 32);
  return v;
}
__device__ __forceinline__ float wmaxf(float v) {
#pragma unroll
  for (int o = 16; o >= 1; o >>= 1) v = fmaxf(v, __shfl_xor(v, o, 32));
  return v;
}

// A-fragment (16x32 f16): lanes<16 -> kb=0, lanes>=16 -> kb=8; halves 0..7 = K kb..kb+7,
// halves 8..15 = K kb+16..kb+23  -> two 16B loads at p and p+16.
__device__ __forceinline__ v16h ldA16(const _Float16* p) {
  union { v16h v; v8h h[2]; } u;
  u.h[0] = *(const v8h*)p;
  u.h[1] = *(const v8h*)(p + 16);
  return u.v;
}
// B-fragment (32x16 f16): lane holds col=lane&15, halves 0..7 = K kb..kb+7, halves 8..15 = K kb+16..kb+23 (kb=(lane>>4)*8).
__device__ __forceinline__ v16h ldB16(const _Float16* p) {
  union { v16h v; v8h h[2]; } u;
  u.h[0] = *(const v8h*)p;
  u.h[1] = *(const v8h*)(p + 16);
  return u.v;
}

// C = A(f16 [M][lda]) x Bt(f16 [colsPad][ldb], row-per-output-col) (+bias,+addsrc,opt GELU).
// 8 waves/block as 4(m) x 2(n); each wave owns a 64x32 tile: 4 A-frags x 2 B-frags,
// 8 WMMA per K-step (1.5 b128 loads per WMMA).
__global__ __launch_bounds__(256) void gemm_wmma(
    const _Float16* __restrict__ A, const _Float16* __restrict__ Bt,
    float* __restrict__ C, const float* __restrict__ bias,
    const float* __restrict__ addsrc,
    int M, int Ncols, int Kpad, int lda, int ldb, int ldc,
    long long sA, long long sB, long long sC, int act)
{
  int z = blockIdx.z;
  A  += (size_t)z * sA;
  Bt += (size_t)z * sB;
  size_t cOff = (size_t)z * sC;

  int lane = threadIdx.x & 31;
  int wave = threadIdx.x >> 5;
  int m0 = blockIdx.y * 256 + (wave >> 1) * 64;
  int n0 = blockIdx.x * 64  + (wave & 1) * 32;

  int r0 = m0 + (lane & 15);
  int r1 = r0 + 16, r2 = r0 + 32, r3 = r0 + 48;
  r0 = r0 < M ? r0 : M - 1;
  r1 = r1 < M ? r1 : M - 1;
  r2 = r2 < M ? r2 : M - 1;
  r3 = r3 < M ? r3 : M - 1;
  int aoff = (lane >> 4) << 3;
  const _Float16* pA0 = A + (size_t)r0 * lda + aoff;
  const _Float16* pA1 = A + (size_t)r1 * lda + aoff;
  const _Float16* pA2 = A + (size_t)r2 * lda + aoff;
  const _Float16* pA3 = A + (size_t)r3 * lda + aoff;
  int c0 = n0 + (lane & 15);
  const _Float16* pB0 = Bt + (size_t)c0 * ldb + ((lane >> 4) << 3);
  const _Float16* pB1 = pB0 + (size_t)16 * ldb;

  v8f acc00 = {}, acc01 = {}, acc10 = {}, acc11 = {};
  v8f acc20 = {}, acc21 = {}, acc30 = {}, acc31 = {};
  for (int k = 0; k < Kpad; k += 32) {
    v16h b0 = ldB16(pB0 + k);
    v16h b1 = ldB16(pB1 + k);
    v16h a0 = ldA16(pA0 + k);
    v16h a1 = ldA16(pA1 + k);
    v16h a2 = ldA16(pA2 + k);
    v16h a3 = ldA16(pA3 + k);
    __builtin_prefetch(pA0 + k + 128, 0, 1);
    __builtin_prefetch(pB0 + k + 128, 0, 1);
    acc00 = __builtin_amdgcn_wmma_f32_16x16x32_f16(false, a0, false, b0, (short)0, acc00, false, false);
    acc01 = __builtin_amdgcn_wmma_f32_16x16x32_f16(false, a0, false, b1, (short)0, acc01, false, false);
    acc10 = __builtin_amdgcn_wmma_f32_16x16x32_f16(false, a1, false, b0, (short)0, acc10, false, false);
    acc11 = __builtin_amdgcn_wmma_f32_16x16x32_f16(false, a1, false, b1, (short)0, acc11, false, false);
    acc20 = __builtin_amdgcn_wmma_f32_16x16x32_f16(false, a2, false, b0, (short)0, acc20, false, false);
    acc21 = __builtin_amdgcn_wmma_f32_16x16x32_f16(false, a2, false, b1, (short)0, acc21, false, false);
    acc30 = __builtin_amdgcn_wmma_f32_16x16x32_f16(false, a3, false, b0, (short)0, acc30, false, false);
    acc31 = __builtin_amdgcn_wmma_f32_16x16x32_f16(false, a3, false, b1, (short)0, acc31, false, false);
  }

  int cr = (lane >> 4) * 8;
  int cc = lane & 15;
#define STORE_TILE(ACC, TM, TN)                                           \
  do {                                                                    \
    int col = n0 + (TN) * 16 + cc;                                        \
    if (col < Ncols) {                                                    \
      float bv = bias ? bias[col] : 0.0f;                                 \
      for (int r = 0; r < 8; ++r) {                                       \
        int row = m0 + (TM) * 16 + cr + r;                                \
        if (row < M) {                                                    \
          float v = (ACC)[r] + bv;                                        \
          size_t o = cOff + (size_t)row * ldc + col;                      \
          if (addsrc) v += addsrc[o];                                     \
          if (act) v = 0.5f * v * (1.0f + erff(v * 0.70710678118f));      \
          C[o] = v;                                                       \
        }                                                                 \
      }                                                                   \
    }                                                                     \
  } while (0)
  STORE_TILE(acc00, 0, 0);
  STORE_TILE(acc01, 0, 1);
  STORE_TILE(acc10, 1, 0);
  STORE_TILE(acc11, 1, 1);
  STORE_TILE(acc20, 2, 0);
  STORE_TILE(acc21, 2, 1);
  STORE_TILE(acc30, 3, 0);
  STORE_TILE(acc31, 3, 1);
#undef STORE_TILE
}

__global__ __launch_bounds__(256) void k_pre_reduce(const float* __restrict__ in, float* __restrict__ ms) {
  int b = blockIdx.x, t = threadIdx.x;
  __shared__ float ssum[256], ssq[256];
  float s = 0.f, q = 0.f;
  const float* base = in + (size_t)b * N0v * (2 * Dm);
  for (int e = t; e < N0v * N0v; e += 256) {
    int i = e / N0v, j = e % N0v;
    float v = log10f(base[(size_t)i * (2 * Dm) + Dm + j] + 1.0f);
    s += v; q += v * v;
  }
  ssum[t] = s; ssq[t] = q; __syncthreads();
  for (int st = 128; st > 0; st >>= 1) {
    if (t < st) { ssum[t] += ssum[t + st]; ssq[t] += ssq[t + st]; }
    __syncthreads();
  }
  if (t == 0) {
    const float n = (float)(N0v * N0v);
    float mean = ssum[0] / n;
    float var = (ssq[0] - n * mean * mean) / (n - 1.0f);
    ms[b * 2 + 0] = mean;
    ms[b * 2 + 1] = sqrtf(fmaxf(var, 0.f)) + 1e-6f;
  }
}

__global__ void k_copy_fc(const float* __restrict__ in, float* __restrict__ X, int total) {
  int tid = blockIdx.x * 256 + threadIdx.x;
  if (tid >= total) return;
  int b = tid / (N0v * Dm), rem = tid % (N0v * Dm);
  int i = rem / Dm, k = rem % Dm;
  X[tid] = in[(size_t)b * N0v * (2 * Dm) + (size_t)i * (2 * Dm) + k];
}

__global__ void k_write_M(const float* __restrict__ in, const float* __restrict__ sc_sel,
                          const float* __restrict__ ms, float* __restrict__ Mm, int total) {
  int tid = blockIdx.x * 256 + threadIdx.x;
  if (tid >= total) return;
  int b = tid / (N0v * N0v), rem = tid % (N0v * N0v);
  int i = rem / N0v, j = rem % N0v;
  float v = log10f(in[(size_t)b * N0v * (2 * Dm) + (size_t)i * (2 * Dm) + Dm + j] + 1.0f);
  v = (v - ms[b * 2]) / ms[b * 2 + 1];
  float g = 1.f / (1.f + expf(-sc_sel[i * N0v + j]));
  Mm[tid] = v * g;
}

__global__ void k_cvt_f16(const float* __restrict__ src, _Float16* __restrict__ dst,
                          int cvalid, int ldi, int ldo, int total) {
  int tid = blockIdx.x * 256 + threadIdx.x;
  if (tid >= total) return;
  int r = tid / ldo, c = tid % ldo;
  dst[tid] = (c < cvalid) ? (_Float16)src[(size_t)r * ldi + c] : (_Float16)0.f;
}

__global__ void k_stage_wt(const float* __restrict__ W, _Float16* __restrict__ Wt,
                           int K, int N, int kpad, int total) {
  int tid = blockIdx.x * 256 + threadIdx.x;
  if (tid >= total) return;
  int n = tid / kpad, k = tid % kpad;
  float v = 0.f;
  if (n < N && k < K) v = W[(size_t)k * N + n];
  Wt[tid] = (_Float16)v;
}

__global__ void k_stage_qk(const float* __restrict__ src, _Float16* __restrict__ dst,
                           int Nv, int rowsPad, int total) {
  int tid = blockIdx.x * 256 + threadIdx.x;
  if (tid >= total) return;
  int z = tid / (rowsPad * DHP), rem = tid % (rowsPad * DHP);
  int n = rem / DHP, d = rem % DHP;
  int b = z / NH, h = z % NH;
  float v = 0.f;
  if (n < Nv && d < DHd) v = src[((size_t)b * Nv + n) * Dm + h * DHd + d];
  dst[tid] = (_Float16)v;
}

__global__ void k_stage_v(const float* __restrict__ src, _Float16* __restrict__ dst,
                          int Nv, int total) {
  int tid = blockIdx.x * 256 + threadIdx.x;
  if (tid >= total) return;
  int z = tid / (DHP * TPAD), rem = tid % (DHP * TPAD);
  int d = rem / TPAD, n = rem % TPAD;
  int b = z / NH, h = z % NH;
  float v = 0.f;
  if (d < DHd && n < Nv) v = src[((size_t)b * Nv + n) * Dm + h * DHd + d];
  dst[tid] = (_Float16)v;
}

__global__ void k_stage_cat(const float* __restrict__ Ohd, _Float16* __restrict__ dst,
                            int Nv, int total) {
  int tid = blockIdx.x * 256 + threadIdx.x;
  if (tid >= total) return;
  int row = tid / TPAD, c = tid % TPAD;
  float v = 0.f;
  if (c < Dm) {
    int b = row / Nv, n = row % Nv;
    int h = c / DHd, d = c % DHd;
    v = Ohd[((size_t)(b * NH + h)) * ((size_t)TPAD * 112) + (size_t)n * 112 + d];
  }
  dst[tid] = (_Float16)v;
}

__global__ __launch_bounds__(256) void k_softmax(const float* __restrict__ S,
    const float* __restrict__ Mm, _Float16* __restrict__ P, int Nv) {
  int lane = threadIdx.x & 31, w = threadIdx.x >> 5;
  int i = blockIdx.x * 8 + w;
  int z = blockIdx.y;
  if (i >= Nv) return;
  int b = z / NH;
  const float* srow = S + (size_t)z * SSTR + (size_t)i * Nv;
  const float* mrow = Mm + ((size_t)b * Nv + i) * Nv;
  float mx = -3.0e38f;
  for (int j = lane; j < Nv; j += 32)
    mx = fmaxf(mx, srow[j] * ATT_SCALE * (1.f + mrow[j]));
  mx = wmaxf(mx);
  float sum = 0.f;
  for (int j = lane; j < Nv; j += 32)
    sum += __expf(srow[j] * ATT_SCALE * (1.f + mrow[j]) - mx);
  sum = wsum(sum);
  float inv = 1.f / sum;
  _Float16* prow = P + (size_t)z * SSTR + (size_t)i * TPAD;
  for (int j = lane; j < TPAD; j += 32) {
    float v = 0.f;
    if (j < Nv) v = __expf(srow[j] * ATT_SCALE * (1.f + mrow[j]) - mx) * inv;
    prow[j] = (_Float16)v;
  }
}

__global__ __launch_bounds__(256) void k_layernorm(const float* __restrict__ F,
    const float* __restrict__ g, const float* __restrict__ bta,
    float* __restrict__ out, int rows) {
  int lane = threadIdx.x & 31, w = threadIdx.x >> 5;
  int row = blockIdx.x * 8 + w;
  if (row >= rows) return;
  const float* f = F + (size_t)row * Dm;
  float s = 0.f, sq = 0.f;
  for (int c = lane; c < Dm; c += 32) { float v = f[c]; s += v; sq += v * v; }
  s = wsum(s); sq = wsum(sq);
  float mean = s / (float)Dm;
  float var = sq / (float)Dm - mean * mean;
  float r = rsqrtf(fmaxf(var, 0.f) + 1e-5f);
  float* o = out + (size_t)row * Dm;
  for (int c = lane; c < Dm; c += 32) o[c] = (f[c] - mean) * r * g[c] + bta[c];
}

__global__ void k_mean_rows(const float* __restrict__ X, float* __restrict__ gf,
                            int Nv, int total) {
  int tid = blockIdx.x * 256 + threadIdx.x;
  if (tid >= total) return;
  int b = tid / Dm, d = tid % Dm;
  float s = 0.f;
  const float* p = X + (size_t)b * Nv * Dm + d;
  for (int n = 0; n < Nv; ++n) s += p[(size_t)n * Dm];
  gf[tid] = s / (float)Nv;
}

__global__ __launch_bounds__(128) void k_classifier(const float* __restrict__ gf,
    const float* __restrict__ r1w, const float* __restrict__ r1b,
    const float* __restrict__ r2w, const float* __restrict__ r2b,
    float* __restrict__ probs_out) {
  int b = blockIdx.x, j = threadIdx.x;
  __shared__ float hbuf[128];
  const float* g = gf + (size_t)b * Dm;
  float a = r1b[j];
  for (int d = 0; d < Dm; ++d) a += g[d] * r1w[(size_t)d * 128 + j];
  hbuf[j] = a > 0.f ? a : 0.01f * a;
  __syncthreads();
  if (j == 0) {
    float l0 = r2b[0], l1 = r2b[1];
    for (int t = 0; t < 128; ++t) { l0 += hbuf[t] * r2w[t * 2 + 0]; l1 += hbuf[t] * r2w[t * 2 + 1]; }
    float mx = fmaxf(l0, l1);
    float e0 = expf(l0 - mx), e1 = expf(l1 - mx), s = e0 + e1;
    probs_out[b * 2 + 0] = e0 / s;
    probs_out[b * 2 + 1] = e1 / s;
  }
}

__global__ void k_pool_scores(const float* __restrict__ X, const float* __restrict__ Mm,
                              const float* __restrict__ pws, const float* __restrict__ pwf,
                              const float* __restrict__ pmw, const float* __restrict__ pmb,
                              float* __restrict__ scores, int total) {
  int tid = blockIdx.x * 256 + threadIdx.x;
  if (tid >= total) return;
  int b = tid / N0v, i = tid % N0v;
  const float* mr = Mm + ((size_t)b * N0v + i) * N0v;
  const float* xr = X + ((size_t)b * N0v + i) * Dm;
  float s1 = 0.f, s2 = 0.f;
  for (int j = 0; j < N0v; ++j) s1 += mr[j] * pws[j];
  for (int k = 0; k < Dm; ++k) s2 += xr[k] * pwf[k];
  float t = fabsf(s1) * pmw[0] + fabsf(s2) * pmw[1] + pmb[0];
  scores[tid] = 1.f / (1.f + expf(-t));
}

__global__ __launch_bounds__(128) void k_pool_select(const float* __restrict__ scores,
                                                     int* __restrict__ idx) {
  int b = blockIdx.x, t = threadIdx.x;
  __shared__ float s[N0v];
  __shared__ int kept[N0v];
  for (int i = t; i < N0v; i += 128) s[i] = scores[b * N0v + i];
  __syncthreads();
  for (int i = t; i < N0v; i += 128) {
    float si = s[i];
    int rank = 0;
    for (int j = 0; j < N0v; ++j) {
      float sj = s[j];
      rank += (sj > si) || (sj == si && j < i);
    }
    kept[i] = rank < N1v ? 1 : 0;
  }
  __syncthreads();
  for (int i = t; i < N0v; i += 128) {
    if (kept[i]) {
      int pos = 0;
      for (int j = 0; j < i; ++j) pos += kept[j];
      if (pos < N1v) idx[b * N1v + pos] = i;
    }
  }
}

__global__ void k_gather_x(const float* __restrict__ Xin, const float* __restrict__ scores,
                           const int* __restrict__ idx, float* __restrict__ Xout, int total) {
  int tid = blockIdx.x * 256 + threadIdx.x;
  if (tid >= total) return;
  int b = tid / (N1v * Dm), rem = tid % (N1v * Dm);
  int p = rem / Dm, c = rem % Dm;
  int i = idx[b * N1v + p];
  Xout[tid] = Xin[((size_t)b * N0v + i) * Dm + c] * scores[b * N0v + i];
}

__global__ void k_gather_m(const float* __restrict__ Min, const int* __restrict__ idx,
                           float* __restrict__ Mout, int total) {
  int tid = blockIdx.x * 256 + threadIdx.x;
  if (tid >= total) return;
  int b = tid / (N1v * N1v), rem = tid % (N1v * N1v);
  int p = rem / N1v, q = rem % N1v;
  int ip = idx[b * N1v + p], iq = idx[b * N1v + q];
  Mout[tid] = Min[((size_t)b * N0v + ip) * N0v + iq];
}

__global__ __launch_bounds__(128) void k_final(const float* __restrict__ X,
                                               const float* __restrict__ probs,
                                               float* __restrict__ out) {
  int b = blockIdx.x, t = threadIdx.x;
  __shared__ float feat[Dm];
  __shared__ float red[128];
  for (int d = t; d < Dm; d += 128) {
    float s = 0.f;
    for (int n = 0; n < N1v; ++n) s += X[((size_t)b * N1v + n) * Dm + d];
    feat[d] = s / (float)N1v;
  }
  __syncthreads();
  float p = 0.f;
  for (int d = t; d < Dm; d += 128) p += feat[d] * feat[d];
  red[t] = p; __syncthreads();
  for (int s2 = 64; s2 > 0; s2 >>= 1) {
    if (t < s2) red[t] += red[t + s2];
    __syncthreads();
  }
  float inv = 1.f / fmaxf(sqrtf(red[0]), 1e-12f);
  for (int d = t; d < Dm; d += 128) out[(size_t)b * Dm + d] = feat[d] * inv;
  if (t < 2) out[(size_t)Bn * Dm + b * 2 + t] =
      0.5f * (probs[b * 2 + t] + probs[Bn * 2 + b * 2 + t]);
}

extern "C" void kernel_launch(void* const* d_in, const int* in_sizes, int n_in,
                              void* d_out, int out_size, void* d_ws, size_t ws_size,
                              hipStream_t stream) {
  (void)in_sizes; (void)n_in; (void)out_size; (void)ws_size;
  const float* inputs  = (const float*)d_in[0];
  const float* sc_sel  = (const float*)d_in[1];
  const float* wq      = (const float*)d_in[2];
  const float* wk      = (const float*)d_in[3];
  const float* wv      = (const float*)d_in[4];
  const float* wo      = (const float*)d_in[5];
  const float* bo      = (const float*)d_in[6];
  const float* ffn_w1  = (const float*)d_in[7];
  const float* ffn_b1  = (const float*)d_in[8];
  const float* ffn_w2  = (const float*)d_in[9];
  const float* ffn_b2  = (const float*)d_in[10];
  const float* ln_g    = (const float*)d_in[11];
  const float* ln_b    = (const float*)d_in[12];
  const float* pool_ws = (const float*)d_in[13];
  const float* pool_wf = (const float*)d_in[14];
  const float* pmlp_w  = (const float*)d_in[15];
  const float* pmlp_b  = (const float*)d_in[16];
  const float* r1w     = (const float*)d_in[17];
  const float* r1b     = (const float*)d_in[18];
  const float* r2w     = (const float*)d_in[19];
  const float* r2b     = (const float*)d_in[20];
  float* out = (float*)d_out;

  char* ws = (char*)d_ws;
  size_t off = 0;
  auto take = [&](size_t bytes) -> void* {
    void* p = ws + off;
    off += (bytes + 255) & ~(size_t)255;
    return p;
  };
  float*    Xc0     = (float*)take((size_t)Bn * N0v * Dm * 4);
  float*    Xc1     = (float*)take((size_t)Bn * N1v * Dm * 4);
  float*    Mc0     = (float*)take((size_t)Bn * N0v * N0v * 4);
  float*    Mc1     = (float*)take((size_t)Bn * N1v * N1v * 4);
  float*    meanstd = (float*)take((size_t)Bn * 2 * 4);
  float*    Qf      = (float*)take((size_t)Bn * N0v * Dm * 4);
  float*    Kf      = (float*)take((size_t)Bn * N0v * Dm * 4);
  float*    Vf      = (float*)take((size_t)Bn * N0v * Dm * 4);
  _Float16* Xh      = (_Float16*)take((size_t)Bn * N0v * TPAD * 2);
  _Float16* Hh      = (_Float16*)take((size_t)Bn * N0v * HIDm * 2);
  _Float16* Wqt     = (_Float16*)take((size_t)448 * TPAD * 2);
  _Float16* Wkt     = (_Float16*)take((size_t)448 * TPAD * 2);
  _Float16* Wvt     = (_Float16*)take((size_t)448 * TPAD * 2);
  _Float16* Wot     = (_Float16*)take((size_t)448 * TPAD * 2);
  _Float16* W1t     = (_Float16*)take((size_t)832 * TPAD * 2);
  _Float16* W2t     = (_Float16*)take((size_t)448 * HIDm * 2);
  _Float16* Qh      = (_Float16*)take((size_t)Bn * NH * TPAD * DHP * 2);
  _Float16* Kh      = (_Float16*)take((size_t)Bn * NH * 448 * DHP * 2);
  _Float16* Vt      = (_Float16*)take((size_t)Bn * NH * DHP * TPAD * 2);
  float*    Sb      = (float*)take((size_t)Bn * NH * SSTR * 4);
  _Float16* Pb      = (_Float16*)take((size_t)Bn * NH * SSTR * 2);
  float*    Ohd     = (float*)take((size_t)Bn * NH * TPAD * 112 * 4);
  _Float16* ACat    = (_Float16*)take((size_t)Bn * N0v * TPAD * 2);
  float*    X1      = (float*)take((size_t)Bn * N0v * Dm * 4);
  float*    Hb      = (float*)take((size_t)Bn * N0v * HIDm * 4);
  float*    Fb      = (float*)take((size_t)Bn * N0v * Dm * 4);
  float*    scoresB = (float*)take((size_t)Bn * N0v * 4);
  int*      idxB    = (int*)take((size_t)Bn * N1v * 4);
  float*    gf      = (float*)take((size_t)Bn * Dm * 4);
  float*    probs   = (float*)take((size_t)2 * Bn * 2 * 4);

  auto blocks = [](long long total) { return (unsigned)((total + 255) / 256); };
  auto gemm = [&](const _Float16* A, const _Float16* Bt, float* C,
                  const float* bias, const float* addsrc,
                  int M, int Ncols, int NcolsPad, int Kpad,
                  int lda, int ldb, int ldc,
                  long long sA, long long sB, long long sC, int batch, int act) {
    dim3 g(NcolsPad / 64, (M + 255) / 256, batch);
    gemm_wmma<<<g, 256, 0, stream>>>(A, Bt, C, bias, addsrc, M, Ncols, Kpad,
                                     lda, ldb, ldc, sA, sB, sC, act);
  };

  // ---- preprocess: split fc/sc, normalize log10(sc+1), M = sc_n * sigmoid(sc_sel) ----
  k_pre_reduce<<<Bn, 256, 0, stream>>>(inputs, meanstd);
  { int t = Bn * N0v * Dm;  k_copy_fc<<<blocks(t), 256, 0, stream>>>(inputs, Xc0, t); }
  { int t = Bn * N0v * N0v; k_write_M<<<blocks(t), 256, 0, stream>>>(inputs, sc_sel, meanstd, Mc0, t); }

  for (int layer = 0; layer < 2; ++layer) {
    const int Nv = (layer == 0) ? N0v : N1v;
    const int NT = ((Nv + 63) / 64) * 64;   // token pad for B-operand rows: 448 / 320
    const int NP = ((Nv + 31) / 32) * 32;   // token pad for K dim: 416 / 288
    const int Mr = Bn * Nv;
    float* X  = (layer == 0) ? Xc0 : Xc1;
    float* Mm = (layer == 0) ? Mc0 : Mc1;
    const float* wq_l = wq + (size_t)layer * Dm * Dm;
    const float* wk_l = wk + (size_t)layer * Dm * Dm;
    const float* wv_l = wv + (size_t)layer * Dm * Dm;
    const float* wo_l = wo + (size_t)layer * Dm * Dm;
    const float* bo_l = bo + (size_t)layer * Dm;
    const float* w1_l = ffn_w1 + (size_t)layer * Dm * HIDm;
    const float* b1_l = ffn_b1 + (size_t)layer * HIDm;
    const float* w2_l = ffn_w2 + (size_t)layer * HIDm * Dm;
    const float* b2_l = ffn_b2 + (size_t)layer * Dm;

    // stage weights transposed to f16
    { int t = 448 * TPAD;
      k_stage_wt<<<blocks(t), 256, 0, stream>>>(wq_l, Wqt, Dm, Dm, TPAD, t);
      k_stage_wt<<<blocks(t), 256, 0, stream>>>(wk_l, Wkt, Dm, Dm, TPAD, t);
      k_stage_wt<<<blocks(t), 256, 0, stream>>>(wv_l, Wvt, Dm, Dm, TPAD, t);
      k_stage_wt<<<blocks(t), 256, 0, stream>>>(wo_l, Wot, Dm, Dm, TPAD, t); }
    { int t = 832 * TPAD; k_stage_wt<<<blocks(t), 256, 0, stream>>>(w1_l, W1t, Dm, HIDm, TPAD, t); }
    { int t = 448 * HIDm; k_stage_wt<<<blocks(t), 256, 0, stream>>>(w2_l, W2t, HIDm, Dm, HIDm, t); }

    // X -> f16 (K padded to 416)
    { int t = Mr * TPAD; k_cvt_f16<<<blocks(t), 256, 0, stream>>>(X, Xh, Dm, Dm, TPAD, t); }

    // Q/K/V projections
    gemm(Xh, Wqt, Qf, nullptr, nullptr, Mr, Dm, 448, TPAD, TPAD, TPAD, Dm, 0, 0, 0, 1, 0);
    gemm(Xh, Wkt, Kf, nullptr, nullptr, Mr, Dm, 448, TPAD, TPAD, TPAD, Dm, 0, 0, 0, 1, 0);
    gemm(Xh, Wvt, Vf, nullptr, nullptr, Mr, Dm, 448, TPAD, TPAD, TPAD, Dm, 0, 0, 0, 1, 0);

    // head split (f16, padded)
    { int t = Bn * NH * TPAD * DHP; k_stage_qk<<<blocks(t), 256, 0, stream>>>(Qf, Qh, Nv, TPAD, t); }
    { int t = Bn * NH * NT * DHP;   k_stage_qk<<<blocks(t), 256, 0, stream>>>(Kf, Kh, Nv, NT, t); }
    { int t = Bn * NH * DHP * TPAD; k_stage_v<<<blocks(t), 256, 0, stream>>>(Vf, Vt, Nv, t); }

    // S = Q @ K^T per (b,h)
    gemm(Qh, Kh, Sb, nullptr, nullptr, Nv, Nv, NT, DHP, DHP, DHP, Nv,
         (long long)TPAD * DHP, (long long)NT * DHP, SSTR, Bn * NH, 0);

    // softmax(scale, mask) -> P (f16, zero-padded)
    { dim3 g((Nv + 7) / 8, Bn * NH); k_softmax<<<g, 256, 0, stream>>>(Sb, Mm, Pb, Nv); }

    // O = P @ V per (b,h)
    gemm(Pb, Vt, Ohd, nullptr, nullptr, Nv, DHd, DHP, NP, TPAD, TPAD, 112,
         SSTR, (long long)DHP * TPAD, (long long)TPAD * 112, Bn * NH, 0);

    // concat heads, out-proj (+bo, +X residual) -> X1
    { int t = Mr * TPAD; k_stage_cat<<<blocks(t), 256, 0, stream>>>(Ohd, ACat, Nv, t); }
    gemm(ACat, Wot, X1, bo_l, X, Mr, Dm, 448, TPAD, TPAD, TPAD, Dm, 0, 0, 0, 1, 0);

    // FFN: Fb = gelu(X1@W1+b1)@W2 + b2 + X1
    { int t = Mr * TPAD; k_cvt_f16<<<blocks(t), 256, 0, stream>>>(X1, Xh, Dm, Dm, TPAD, t); }
    gemm(Xh, W1t, Hb, b1_l, nullptr, Mr, HIDm, 832, TPAD, TPAD, TPAD, HIDm, 0, 0, 0, 1, 1);
    { int t = Mr * HIDm; k_cvt_f16<<<blocks(t), 256, 0, stream>>>(Hb, Hh, HIDm, HIDm, HIDm, t); }
    gemm(Hh, W2t, Fb, b2_l, X1, Mr, Dm, 448, HIDm, HIDm, HIDm, Dm, 0, 0, 0, 1, 0);

    // LayerNorm -> X (in place, compact)
    { dim3 g((Mr + 7) / 8); k_layernorm<<<g, 256, 0, stream>>>(Fb, ln_g + (size_t)layer * Dm,
                                                               ln_b + (size_t)layer * Dm, X, Mr); }

    // classifier head for this depth
    { int t = Bn * Dm; k_mean_rows<<<blocks(t), 256, 0, stream>>>(X, gf, Nv, t); }
    k_classifier<<<Bn, 128, 0, stream>>>(gf, r1w + (size_t)layer * Dm * 128,
                                         r1b + (size_t)layer * 128,
                                         r2w + (size_t)layer * 128 * 2,
                                         r2b + (size_t)layer * 2,
                                         probs + (size_t)layer * Bn * 2);

    // pooling (after layer 0 only)
    if (layer == 0) {
      { int t = Bn * N0v;
        k_pool_scores<<<blocks(t), 256, 0, stream>>>(X, Mm, pool_ws, pool_wf, pmlp_w, pmlp_b, scoresB, t); }
      k_pool_select<<<Bn, 128, 0, stream>>>(scoresB, idxB);
      { int t = Bn * N1v * Dm;  k_gather_x<<<blocks(t), 256, 0, stream>>>(X, scoresB, idxB, Xc1, t); }
      { int t = Bn * N1v * N1v; k_gather_m<<<blocks(t), 256, 0, stream>>>(Mm, idxB, Mc1, t); }
    }
  }

  // final: normalized feature + averaged probs -> d_out
  k_final<<<Bn, 128, 0, stream>>>(Xc1, probs, out);
}
